// CoeffNet_28071906246845
// MI455X (gfx1250) — matrix operand order repaired and checked by reading, not compiled, not run on hardware
//
#include <hip/hip_runtime.h>
#include <math.h>

typedef __attribute__((ext_vector_type(16))) _Float16 v16h;
typedef __attribute__((ext_vector_type(8)))  float    v8f;

#define N_NODES_C 20000
#define N_EDGES_C 320000
#define F_DIM 64
#define K_DIM 32
#define R_MAX_C 2.5f
#define WAVES_PER_BLOCK 8

// ---------------------------------------------------------------------------
// Kernel 1: x = tensor_dense(x_dftb, W_in0, W_in1, b_in)
// One block of 64 threads per (node, component) row; row broadcast via LDS.
// ---------------------------------------------------------------------------
__global__ __launch_bounds__(F_DIM)
void k_input(const float* __restrict__ x_dftb,
             const float* __restrict__ W_in0,
             const float* __restrict__ W_in1,
             const float* __restrict__ b_in,
             float* __restrict__ x_out) {
    __shared__ float row[F_DIM];
    const int r = blockIdx.x;        // node*4 + l
    const int l = r & 3;
    const int f = threadIdx.x;
    row[f] = x_dftb[(size_t)r * F_DIM + f];
    __syncthreads();
    const float* __restrict__ W = (l == 0) ? W_in0 : W_in1;
    float acc = (l == 0) ? b_in[f] : 0.0f;
#pragma unroll 8
    for (int k = 0; k < F_DIM; ++k) acc += row[k] * W[k * F_DIM + f];
    x_out[(size_t)r * F_DIM + f] = acc;
}

// ---------------------------------------------------------------------------
// Kernel 2 (per round): fused edge kernel.
//   Wave handles 16 edges: RBF -> WMMA (16x32 @ 32x64, f16 in / f32 acc)
//   -> LDS transpose -> per-(edge,feature) message + atomic scatter-add.
// ---------------------------------------------------------------------------
__global__ __launch_bounds__(32 * WAVES_PER_BLOCK)
void k_edge(const float* __restrict__ coords,
            const int*   __restrict__ dst_idx,
            const int*   __restrict__ src_idx,
            const float* __restrict__ W_basis,
            const float* __restrict__ b_basis,
            const float* __restrict__ path_w,
            const float* __restrict__ x_cur,
            float*       __restrict__ x_next,
            int round_i) {
    __shared__ float s_t[WAVES_PER_BLOCK][16 * F_DIM];   // 32 KB: t tiles
    __shared__ float s_rhat[WAVES_PER_BLOCK][16][3];
    __shared__ int   s_src[WAVES_PER_BLOCK][16];
    __shared__ int   s_dst[WAVES_PER_BLOCK][16];
    __shared__ float s_bb[F_DIM];
    __shared__ float s_p[5][F_DIM];

    const int tid = threadIdx.x;
    if (tid < F_DIM) s_bb[tid] = b_basis[round_i * F_DIM + tid];
    for (int q = tid; q < 5 * F_DIM; q += 32 * WAVES_PER_BLOCK)
        s_p[q >> 6][q & (F_DIM - 1)] = path_w[round_i * 5 * F_DIM + q];
    __syncthreads();

    const int wave = tid >> 5;
    const int lane = tid & 31;
    const int le   = lane & 15;           // local edge 0..15
    const int hi   = (lane >> 4) & 1;     // half-wave selector

    const int tile = blockIdx.x * WAVES_PER_BLOCK + wave;
    if (tile * 16 >= N_EDGES_C) return;   // wave-uniform: EXEC stays all-ones
    const int e = tile * 16 + le;

    // ---- per-edge geometry (lanes L and L+16 redundantly compute edge L) ----
    const int sn = src_idx[e];
    const int dn = dst_idx[e];
    const float rx = coords[dn * 3 + 0] - coords[sn * 3 + 0];
    const float ry = coords[dn * 3 + 1] - coords[sn * 3 + 1];
    const float rz = coords[dn * 3 + 2] - coords[sn * 3 + 2];
    const float d  = sqrtf(rx * rx + ry * ry + rz * rz + 1e-12f);
    const float inv = 1.0f / d;
    const float hx = rx * inv, hy = ry * inv, hz = rz * inv;
    if (hi == 0) {
        s_rhat[wave][le][0] = hx;
        s_rhat[wave][le][1] = hy;
        s_rhat[wave][le][2] = hz;
        s_src[wave][le] = sn;
        s_dst[wave][le] = dn;
    }

    // ---- A fragment: 16 RBF values per lane (16-bit A layout, 16x32) ----
    const float width = R_MAX_C / (float)(K_DIM - 1);
    const float invw  = 1.0f / width;
    v16h a_frag;
#pragma unroll
    for (int h = 0; h < 16; ++h) {
        const int k = (h < 8 ? h : h + 8) + (hi ? 8 : 0);
        const float u = (d - width * (float)k) * invw;
        a_frag[h] = (_Float16)__expf(-u * u);
    }

    // ---- B fragments: W_basis[round_i] (32x64) as four 32x16 tiles ----
    const float* __restrict__ Wb = W_basis + (size_t)round_i * K_DIM * F_DIM;
    v16h b_frag[4];
#pragma unroll
    for (int j = 0; j < 4; ++j) {
        const int n = j * 16 + le;
#pragma unroll
        for (int h = 0; h < 16; ++h) {
            const int k = (h < 8 ? h : h + 8) + (hi ? 8 : 0);
            b_frag[j][h] = (_Float16)Wb[k * F_DIM + n];
        }
    }

    // ---- t(16x64) = rbf(16x32) @ Wb(32x64), f32 accumulate ----
    v8f acc[4];
#pragma unroll
    for (int j = 0; j < 4; ++j) {
#pragma unroll
        for (int v = 0; v < 8; ++v) acc[j][v] = 0.0f;
        acc[j] = __builtin_amdgcn_wmma_f32_16x16x32_f16(
            false, a_frag, false, b_frag[j], (short)0, acc[j], false, false);
    }

    // ---- spill C-layout to LDS: lane c holds (M=v [+8 if hi], N=16j+c) ----
#pragma unroll
    for (int j = 0; j < 4; ++j) {
#pragma unroll
        for (int v = 0; v < 8; ++v) {
            const int M = v + hi * 8;
            const int N = j * 16 + le;
            s_t[wave][M * F_DIM + N] = acc[j][v];
        }
    }
    __builtin_amdgcn_wave_barrier();
    asm volatile("s_wait_dscnt 0" ::: "memory");   // in-wave DS ordering

    // ---- message phase: 16 edges x 64 features = 1024 items, 32 lanes ----
#pragma unroll 1
    for (int it = 0; it < 32; ++it) {
        const int item = it * 32 + lane;
        const int f = item & (F_DIM - 1);
        const int m = item >> 6;                    // local edge
        const float t  = s_t[wave][m * F_DIM + f];
        const float ux = s_rhat[wave][m][0];
        const float uy = s_rhat[wave][m][1];
        const float uz = s_rhat[wave][m][2];
        const int   sN = s_src[wave][m];
        const int   dN = s_dst[wave][m];
        const float g0 = t + s_bb[f];

        const float* __restrict__ xa = x_cur + (size_t)sN * 4 * F_DIM + f;
        const float a0 = xa[0];
        const float a1 = xa[F_DIM];
        const float a2 = xa[2 * F_DIM];
        const float a3 = xa[3 * F_DIM];

        const float avdot = a1 * ux + a2 * uy + a3 * uz;
        const float cx = a2 * uz - a3 * uy;
        const float cy = a3 * ux - a1 * uz;
        const float cz = a1 * uy - a2 * ux;

        const float p0 = s_p[0][f], p1 = s_p[1][f], p2 = s_p[2][f];
        const float p3 = s_p[3][f], p4 = s_p[4][f];

        const float m0 = p0 * a0 * g0 + p1 * avdot * t;
        const float m1 = p2 * a0 * t * ux + p3 * a1 * g0 + p4 * t * cx;
        const float m2 = p2 * a0 * t * uy + p3 * a2 * g0 + p4 * t * cy;
        const float m3 = p2 * a0 * t * uz + p3 * a3 * g0 + p4 * t * cz;

        float* __restrict__ xo = x_next + (size_t)dN * 4 * F_DIM + f;
        atomicAdd(xo,             m0);
        atomicAdd(xo + F_DIM,     m1);
        atomicAdd(xo + 2 * F_DIM, m2);
        atomicAdd(xo + 3 * F_DIM, m3);
    }
}

// ---------------------------------------------------------------------------
// Kernel 3: out = tensor_dense(x, W_out0, W_out1, b_out) -> (N,4,1)
// One wave per (node, component) row; shuffle reduction.
// ---------------------------------------------------------------------------
__global__ __launch_bounds__(32 * WAVES_PER_BLOCK)
void k_output(const float* __restrict__ x,
              const float* __restrict__ W_out0,
              const float* __restrict__ W_out1,
              const float* __restrict__ b_out,
              float* __restrict__ out) {
    const int wave = threadIdx.x >> 5;
    const int lane = threadIdx.x & 31;
    const int row  = blockIdx.x * WAVES_PER_BLOCK + wave;
    if (row >= N_NODES_C * 4) return;
    const int l = row & 3;
    const float* __restrict__ W = (l == 0) ? W_out0 : W_out1;
    float v = x[(size_t)row * F_DIM + lane]      * W[lane]
            + x[(size_t)row * F_DIM + 32 + lane] * W[32 + lane];
#pragma unroll
    for (int off = 16; off > 0; off >>= 1) v += __shfl_down(v, off, 32);
    if (lane == 0) out[row] = v + ((l == 0) ? b_out[0] : 0.0f);
}

// ---------------------------------------------------------------------------
extern "C" void kernel_launch(void* const* d_in, const int* in_sizes, int n_in,
                              void* d_out, int out_size, void* d_ws, size_t ws_size,
                              hipStream_t stream) {
    const float* x_dftb  = (const float*)d_in[0];
    const float* coords  = (const float*)d_in[1];
    const int*   dst_idx = (const int*)  d_in[2];
    const int*   src_idx = (const int*)  d_in[3];
    const float* W_in0   = (const float*)d_in[4];
    const float* W_in1   = (const float*)d_in[5];
    const float* b_in    = (const float*)d_in[6];
    const float* W_basis = (const float*)d_in[7];
    const float* b_basis = (const float*)d_in[8];
    const float* path_w  = (const float*)d_in[9];
    const float* W_out0  = (const float*)d_in[10];
    const float* W_out1  = (const float*)d_in[11];
    const float* b_out   = (const float*)d_in[12];

    const size_t xbytes = (size_t)N_NODES_C * 4 * F_DIM * sizeof(float);
    float* bufA = (float*)d_ws;
    float* bufB = (float*)((char*)d_ws + xbytes);

    // x = tensor_dense(x_dftb, W_in0, W_in1, b_in)
    k_input<<<N_NODES_C * 4, F_DIM, 0, stream>>>(x_dftb, W_in0, W_in1, b_in, bufA);

    // 3 message-passing rounds, ping-pong buffers (x_next starts as copy of x_cur)
    float* cur = bufA;
    float* nxt = bufB;
    const int edge_blocks = (N_EDGES_C / 16 + WAVES_PER_BLOCK - 1) / WAVES_PER_BLOCK;
    for (int i = 0; i < 3; ++i) {
        hipMemcpyAsync(nxt, cur, xbytes, hipMemcpyDeviceToDevice, stream);
        k_edge<<<edge_blocks, 32 * WAVES_PER_BLOCK, 0, stream>>>(
            coords, dst_idx, src_idx, W_basis, b_basis, path_w, cur, nxt, i);
        float* tmp = cur; cur = nxt; nxt = tmp;
    }

    // out = tensor_dense(x, W_out0, W_out1, b_out)
    const int out_blocks = (N_NODES_C * 4 + WAVES_PER_BLOCK - 1) / WAVES_PER_BLOCK;
    k_output<<<out_blocks, 32 * WAVES_PER_BLOCK, 0, stream>>>(
        cur, W_out0, W_out1, b_out, (float*)d_out);
}
